// MultiHeadAttention_32581621907512
// MI455X (gfx1250) — compile-verified, matrix operand order checked by reading, not run on hardware
//
#include <hip/hip_runtime.h>
#include <hip/hip_bf16.h>
#include <math.h>

// ---------------------------------------------------------------------------
// Problem constants (match reference)
// ---------------------------------------------------------------------------
#define B_   4
#define L_   1024
#define D_   1024
#define H_   16
#define DK_  64
#define DV_  64
#define HID_ 4096
#define EPS_ 1e-5f

typedef __bf16 bf16_t;
typedef bf16_t v16bf __attribute__((ext_vector_type(16)));
typedef float  v8f   __attribute__((ext_vector_type(8)));
typedef int    v4i   __attribute__((ext_vector_type(4)));

__device__ __forceinline__ unsigned short f2bf(float f) {
    unsigned u = __builtin_bit_cast(unsigned, f);
    return (unsigned short)((u + 0x7FFFu + ((u >> 16) & 1u)) >> 16);   // RNE
}

union FragBF {
    v16bf v;
    uint4 q[2];
    unsigned short s[16];
};

// ---------------------------------------------------------------------------
// CDNA5 async global->LDS staging (ASYNCcnt path), with sync fallback.
// Builtin signature (from compiler diagnostic): param0 = v4i addrspace(1)*,
// so call is (global_src, lds_dst, imm_offset, cpol).
// ---------------------------------------------------------------------------
#if defined(__gfx1250__) && __has_builtin(__builtin_amdgcn_global_load_async_to_lds_b128)
#define HAVE_ASYNC_LDS 1
#else
#define HAVE_ASYNC_LDS 0
#endif

typedef __attribute__((address_space(1))) v4i* gptr_v4i;
typedef __attribute__((address_space(3))) v4i* lptr_v4i;

__device__ __forceinline__ void stage16(const unsigned short* g, unsigned short* l) {
#if HAVE_ASYNC_LDS
    // 16B per lane, memory -> LDS directly, no VGPR round trip
    __builtin_amdgcn_global_load_async_to_lds_b128((gptr_v4i)g, (lptr_v4i)l, 0, 0);
#else
    *(uint4*)l = *(const uint4*)g;
#endif
}

__device__ __forceinline__ void wait_async0() {
#if HAVE_ASYNC_LDS
#if __has_builtin(__builtin_amdgcn_s_wait_asynccnt)
    __builtin_amdgcn_s_wait_asynccnt(0);
#else
    asm volatile("s_wait_asynccnt 0x0" ::: "memory");
#endif
#endif
}

// ---------------------------------------------------------------------------
// fp32 -> bf16 bulk conversion (weights, k, v)
// ---------------------------------------------------------------------------
__global__ void cvt_f32_bf16(const float* __restrict__ in,
                             unsigned short* __restrict__ out, long long n) {
    long long i = (long long)blockIdx.x * blockDim.x + threadIdx.x;
    long long stride = (long long)gridDim.x * blockDim.x;
    for (; i < n; i += stride) out[i] = f2bf(in[i]);
}

// ---------------------------------------------------------------------------
// Self-modulated LayerNorm: out = g*z*LN(x) + b*z
// One 256-thread block per row of D_=1024. wave32 shfl + LDS reduction.
// ---------------------------------------------------------------------------
template <bool WRITE_F32>
__global__ void sln_kernel(const float* __restrict__ x, const float* __restrict__ z,
                           const float* __restrict__ lnw, const float* __restrict__ lnb,
                           const float* __restrict__ g, const float* __restrict__ bb,
                           float* __restrict__ res_out,
                           unsigned short* __restrict__ bf_out) {
    const long long base = (long long)blockIdx.x * D_;
    const int tid = threadIdx.x;
    float v0[4];
    float s = 0.f, ss = 0.f;
#pragma unroll
    for (int j = 0; j < 4; ++j) {
        int c = tid + j * 256;
        float val = x[base + c];
        v0[j] = val; s += val; ss += val * val;
    }
#pragma unroll
    for (int o = 16; o > 0; o >>= 1) {
        s  += __shfl_xor(s,  o, 32);
        ss += __shfl_xor(ss, o, 32);
    }
    __shared__ float red[2][8];
    if ((tid & 31) == 0) { red[0][tid >> 5] = s; red[1][tid >> 5] = ss; }
    __syncthreads();
    float ts = 0.f, tss = 0.f;
#pragma unroll
    for (int w = 0; w < 8; ++w) { ts += red[0][w]; tss += red[1][w]; }
    const float mean = ts * (1.0f / D_);
    const float var  = tss * (1.0f / D_) - mean * mean;
    const float rstd = rsqrtf(var + EPS_);
    const float g0 = g[0], b0 = bb[0];
#pragma unroll
    for (int j = 0; j < 4; ++j) {
        int c = tid + j * 256;
        float zz = z[base + c];
        float ln = (v0[j] - mean) * rstd * lnw[c] + lnb[c];
        float o  = g0 * zz * ln + b0 * zz;
        if (WRITE_F32) res_out[base + c] = o;
        bf_out[base + c] = f2bf(o);
    }
}

// ---------------------------------------------------------------------------
// Row softmax over 1024 fp32, in place. One 256-thread block per row.
// ---------------------------------------------------------------------------
__global__ void softmax_rows(float* __restrict__ p) {
    float* r = p + (long long)blockIdx.x * L_;
    const int tid = threadIdx.x;
    __shared__ float red[8];
    float v[4];
    float mx = -3.4e38f;
#pragma unroll
    for (int j = 0; j < 4; ++j) { v[j] = r[tid + j * 256]; mx = fmaxf(mx, v[j]); }
#pragma unroll
    for (int o = 16; o > 0; o >>= 1) mx = fmaxf(mx, __shfl_xor(mx, o, 32));
    if ((tid & 31) == 0) red[tid >> 5] = mx;
    __syncthreads();
    float m2 = red[0];
#pragma unroll
    for (int w = 1; w < 8; ++w) m2 = fmaxf(m2, red[w]);
    __syncthreads();
    float sum = 0.f;
#pragma unroll
    for (int j = 0; j < 4; ++j) { v[j] = __expf(v[j] - m2); sum += v[j]; }
#pragma unroll
    for (int o = 16; o > 0; o >>= 1) sum += __shfl_xor(sum, o, 32);
    if ((tid & 31) == 0) red[tid >> 5] = sum;
    __syncthreads();
    float tsum = 0.f;
#pragma unroll
    for (int w = 0; w < 8; ++w) tsum += red[w];
    const float inv = 1.0f / tsum;
#pragma unroll
    for (int j = 0; j < 4; ++j) r[tid + j * 256] = v[j] * inv;
}

// ---------------------------------------------------------------------------
// Generic batched WMMA GEMM, bf16 x bf16 -> fp32 accum.
//   C[z][M,N] = act( alpha * A[z][M,K] @ opB(B[z]) + bias ) (+ res)
// Tile: BM x BN per block, BK=32. Wave tile: 64x32 = 4x2 WMMA accumulators.
// NTHREADS = (BM/64)*(BN/32)*32.
// TRANS_B : B is [N,K] row-major (QK^T scores) -> row-contiguous staging.
// A_F32   : A is fp32 in memory, converted to bf16 while staging (sync path).
// Async global->LDS (ASYNCcnt) used for all row-contiguous bf16 staging.
// Per-z operand offset: (z / zdim2)*z1 + (z % zdim2)*z2.
// ---------------------------------------------------------------------------
template <int BM, int BN, int NTHREADS,
          bool TRANS_B, bool A_F32, bool HAS_BIAS, bool DO_GELU, bool HAS_RES, bool OUT_BF16>
__global__ __launch_bounds__(NTHREADS) void gemm_wmma(
    const void* __restrict__ Ap, int lda, long long az1, long long az2,
    const void* __restrict__ Bp, int ldb, long long bz1, long long bz2,
    void*       __restrict__ Cp, int ldc, long long cz1, long long cz2,
    int zdim2,
    const float* __restrict__ bias,
    const float* __restrict__ res,
    int K, float alpha) {
    __shared__ __align__(16) unsigned short As[BM][32];
    __shared__ __align__(16) unsigned short Bs[BN][32];   // [n][k]: K-innermost

    constexpr int WMT = 4;                 // 4 m-tiles of 16 per wave (WM=64)
    constexpr int WNT = 2;                 // 2 n-tiles of 16 per wave (WN=32)
    constexpr int NWC = BN / 32;           // wave columns
    constexpr bool ASYNC_A = !A_F32;
    constexpr bool ASYNC_B = TRANS_B;

    const int z = blockIdx.z;
    const long long aoff = (long long)(z / zdim2) * az1 + (long long)(z % zdim2) * az2;
    const long long boff = (long long)(z / zdim2) * bz1 + (long long)(z % zdim2) * bz2;
    const long long coff = (long long)(z / zdim2) * cz1 + (long long)(z % zdim2) * cz2;

    const int bm = blockIdx.y * BM;
    const int bn = blockIdx.x * BN;
    const int tid  = threadIdx.x;
    const int lane = tid & 31;
    const int wave = tid >> 5;
    const int wrow = (wave / NWC) * 64;
    const int wcol = (wave % NWC) * 32;
    const int lm = lane & 15;       // row (A) / col (B) / N (C)
    const int lh = lane >> 4;       // K-half selector

    v8f acc[WMT][WNT];
#pragma unroll
    for (int mt = 0; mt < WMT; ++mt)
#pragma unroll
        for (int nt = 0; nt < WNT; ++nt)
#pragma unroll
            for (int i = 0; i < 8; ++i) acc[mt][nt][i] = 0.f;

    const unsigned short* A16  = (const unsigned short*)Ap;
    const float*          A32  = (const float*)Ap;
    const unsigned short* Bmat = (const unsigned short*)Bp;

    for (int k0 = 0; k0 < K; k0 += 32) {
        // ---- stage A tile: rows bm..bm+BM-1, K cols k0..k0+31 ----
#pragma unroll
        for (int s = tid; s < BM * 2; s += NTHREADS) {
            const int row  = s >> 1;
            const int kseg = (s & 1) * 16;
            long long gb = aoff + (long long)(bm + row) * lda + k0 + kseg;
            if (A_F32) {
                const float* p = A32 + gb;
                unsigned short tmp[16];
#pragma unroll
                for (int j = 0; j < 16; ++j) tmp[j] = f2bf(p[j]);
                *(uint4*)&As[row][kseg]     = *(const uint4*)&tmp[0];
                *(uint4*)&As[row][kseg + 8] = *(const uint4*)&tmp[8];
            } else if (ASYNC_A) {
                stage16(A16 + gb,     &As[row][kseg]);
                stage16(A16 + gb + 8, &As[row][kseg + 8]);
            }
        }
        // ---- stage B tile into [n][k] layout ----
        if (TRANS_B) {
            // B is [N,K]: already K-innermost, row-contiguous -> async
#pragma unroll
            for (int s = tid; s < BN * 2; s += NTHREADS) {
                const int row  = s >> 1;
                const int kseg = (s & 1) * 16;
                long long gb = boff + (long long)(bn + row) * ldb + k0 + kseg;
                if (ASYNC_B) {
                    stage16(Bmat + gb,     &Bs[row][kseg]);
                    stage16(Bmat + gb + 8, &Bs[row][kseg + 8]);
                }
            }
        } else {
            // B is [K,N]: transpose while storing (VGPR round trip required)
#pragma unroll
            for (int s = tid; s < BN * 2; s += NTHREADS) {
                const int kk   = s / (BN / 16);
                const int nseg = (s % (BN / 16)) * 16;
                long long gb = boff + (long long)(k0 + kk) * ldb + bn + nseg;
                uint4 d0 = *(const uint4*)(Bmat + gb);
                uint4 d1 = *(const uint4*)(Bmat + gb + 8);
                const unsigned short* t0 = (const unsigned short*)&d0;
                const unsigned short* t1 = (const unsigned short*)&d1;
#pragma unroll
                for (int j = 0; j < 8; ++j) {
                    Bs[nseg + j][kk]     = t0[j];
                    Bs[nseg + 8 + j][kk] = t1[j];
                }
            }
        }
        if (ASYNC_A || ASYNC_B) wait_async0();
        __syncthreads();

        // ---- fragments: ISA 16-bit layout, K = lh*8+[0..7] then 16+lh*8+[0..7]
        FragBF af[WMT], bfr[WNT];
#pragma unroll
        for (int mt = 0; mt < WMT; ++mt) {
            int r = wrow + mt * 16 + lm;
            af[mt].q[0] = *(const uint4*)&As[r][lh * 8];
            af[mt].q[1] = *(const uint4*)&As[r][16 + lh * 8];
        }
#pragma unroll
        for (int nt = 0; nt < WNT; ++nt) {
            int c = wcol + nt * 16 + lm;
            bfr[nt].q[0] = *(const uint4*)&Bs[c][lh * 8];
            bfr[nt].q[1] = *(const uint4*)&Bs[c][16 + lh * 8];
        }
#pragma unroll
        for (int mt = 0; mt < WMT; ++mt)
#pragma unroll
            for (int nt = 0; nt < WNT; ++nt)
                acc[mt][nt] = __builtin_amdgcn_wmma_f32_16x16x32_bf16(
                    false, af[mt].v, false, bfr[nt].v,
                    (short)0, acc[mt][nt], false, false);
        __syncthreads();
    }

    // ---- epilogue: C layout VGPR i -> m = i + 8*lh, n = lm ----
    float*          Cf = (float*)Cp;
    unsigned short* Cb = (unsigned short*)Cp;
#pragma unroll
    for (int mt = 0; mt < WMT; ++mt) {
#pragma unroll
        for (int nt = 0; nt < WNT; ++nt) {
            const int gc = bn + wcol + nt * 16 + lm;
            const float bv = HAS_BIAS ? bias[gc] : 0.f;
#pragma unroll
            for (int i = 0; i < 8; ++i) {
                const int gr = bm + wrow + mt * 16 + i + 8 * lh;
                float vv = acc[mt][nt][i] * alpha + bv;
                if (DO_GELU) vv = 0.5f * vv * (1.0f + erff(vv * 0.70710678118654752f));
                const long long ci = coff + (long long)gr * ldc + gc;
                if (HAS_RES) vv += res[ci];
                if (OUT_BF16) Cb[ci] = f2bf(vv);
                else          Cf[ci] = vv;
            }
        }
    }
}

// ---------------------------------------------------------------------------
// Host orchestration
// ---------------------------------------------------------------------------
extern "C" void kernel_launch(void* const* d_in, const int* in_sizes, int n_in,
                              void* d_out, int out_size, void* d_ws, size_t ws_size,
                              hipStream_t stream) {
    (void)in_sizes; (void)n_in; (void)out_size; (void)ws_size;

    const float* q      = (const float*)d_in[0];
    const float* k      = (const float*)d_in[1];
    const float* v      = (const float*)d_in[2];
    const float* z      = (const float*)d_in[3];
    const float* wq_w   = (const float*)d_in[4];
    const float* wq_b   = (const float*)d_in[5];
    const float* wk_w   = (const float*)d_in[6];
    const float* wk_b   = (const float*)d_in[7];
    const float* wv_w   = (const float*)d_in[8];
    const float* wv_b   = (const float*)d_in[9];
    const float* fc_w   = (const float*)d_in[10];
    const float* fc_b   = (const float*)d_in[11];
    const float* mlp1_w = (const float*)d_in[12];
    const float* mlp1_b = (const float*)d_in[13];
    const float* mlp2_w = (const float*)d_in[14];
    const float* mlp2_b = (const float*)d_in[15];
    const float* ln1_w  = (const float*)d_in[16];
    const float* ln1_b  = (const float*)d_in[17];
    const float* g1     = (const float*)d_in[18];
    const float* b1     = (const float*)d_in[19];
    const float* ln2_w  = (const float*)d_in[20];
    const float* ln2_b  = (const float*)d_in[21];
    const float* g2     = (const float*)d_in[22];
    const float* b2     = (const float*)d_in[23];

    const long long ND  = (long long)B_ * L_ * D_;       // 4M elems
    const long long NW  = (long long)D_ * H_ * DK_;      // 1M elems
    const long long NM1 = (long long)D_ * HID_;          // 4M elems
    const long long NH  = (long long)B_ * L_ * HID_;     // 16M elems

    char* wsp = (char*)d_ws;
    size_t off = 0;
    auto carve = [&](size_t bytes) -> char* {
        char* p = wsp + off;
        off += (bytes + 255) & ~(size_t)255;
        return p;
    };
    unsigned short* k_bf    = (unsigned short*)carve(ND * 2);
    unsigned short* v_bf    = (unsigned short*)carve(ND * 2);
    unsigned short* wq_bf   = (unsigned short*)carve(NW * 2);
    unsigned short* wk_bf   = (unsigned short*)carve(NW * 2);
    unsigned short* wv_bf   = (unsigned short*)carve(NW * 2);
    unsigned short* fc_bf   = (unsigned short*)carve(NW * 2);
    unsigned short* mlp1_bf = (unsigned short*)carve(NM1 * 2);
    unsigned short* mlp2_bf = (unsigned short*)carve(NM1 * 2);
    float*          q_res   = (float*)carve(ND * 4);     // residual (post-SLN q)
    unsigned short* q_bf    = (unsigned short*)carve(ND * 2);
    unsigned short* qh_bf   = (unsigned short*)carve(ND * 2);
    unsigned short* kh_bf   = (unsigned short*)carve(ND * 2);
    unsigned short* vh_bf   = (unsigned short*)carve(ND * 2);
    unsigned short* ao_bf   = (unsigned short*)carve(ND * 2);  // attn @ V
    float*          out1    = (float*)carve(ND * 4);     // fc out + residual
    unsigned short* h2_bf   = (unsigned short*)carve(ND * 2);  // SLN2 out
    unsigned short* hid_bf  = (unsigned short*)carve(NH * 2);  // gelu(mlp1)

    float* out_f = (float*)d_out;           // [B,L,D]
    float* attn  = out_f + ND;              // [H*B, L, L] fp32

    // 1) bf16 conversions
    cvt_f32_bf16<<<2048, 256, 0, stream>>>(k,      k_bf,    ND);
    cvt_f32_bf16<<<2048, 256, 0, stream>>>(v,      v_bf,    ND);
    cvt_f32_bf16<<<1024, 256, 0, stream>>>(wq_w,   wq_bf,   NW);
    cvt_f32_bf16<<<1024, 256, 0, stream>>>(wk_w,   wk_bf,   NW);
    cvt_f32_bf16<<<1024, 256, 0, stream>>>(wv_w,   wv_bf,   NW);
    cvt_f32_bf16<<<1024, 256, 0, stream>>>(fc_w,   fc_bf,   NW);
    cvt_f32_bf16<<<2048, 256, 0, stream>>>(mlp1_w, mlp1_bf, NM1);
    cvt_f32_bf16<<<2048, 256, 0, stream>>>(mlp2_w, mlp2_bf, NM1);

    // 2) SLN1: q -> q_res (fp32 residual) + q_bf
    sln_kernel<true><<<B_ * L_, 256, 0, stream>>>(q, z, ln1_w, ln1_b, g1, b1,
                                                  q_res, q_bf);

    // 3) projections: [4096,1024] @ [1024,1024] + bias -> bf16
    gemm_wmma<128, 128, 256, false, false, true, false, false, true>
        <<<dim3(8, 32, 1), 256, 0, stream>>>(
        q_bf, D_, 0, 0, wq_bf, H_ * DK_, 0, 0, qh_bf, H_ * DK_, 0, 0,
        1, wq_b, nullptr, D_, 1.0f);
    gemm_wmma<128, 128, 256, false, false, true, false, false, true>
        <<<dim3(8, 32, 1), 256, 0, stream>>>(
        k_bf, D_, 0, 0, wk_bf, H_ * DK_, 0, 0, kh_bf, H_ * DK_, 0, 0,
        1, wk_b, nullptr, D_, 1.0f);
    gemm_wmma<128, 128, 256, false, false, true, false, false, true>
        <<<dim3(8, 32, 1), 256, 0, stream>>>(
        v_bf, D_, 0, 0, wv_bf, H_ * DV_, 0, 0, vh_bf, H_ * DV_, 0, 0,
        1, wv_b, nullptr, D_, 1.0f);

    // 4) scores: per (h,b): qh_h [L,64] @ kh_h^T -> attn fp32, alpha=1/sqrt(64)
    //    z = h*B + b; operand offset = (z/B)*z1 + (z%B)*z2
    gemm_wmma<128, 128, 256, true, false, false, false, false, false>
        <<<dim3(8, 8, H_ * B_), 256, 0, stream>>>(
        qh_bf, H_ * DK_, (long long)DK_, (long long)L_ * H_ * DK_,
        kh_bf, H_ * DK_, (long long)DK_, (long long)L_ * H_ * DK_,
        attn,  L_,       (long long)B_ * L_ * L_, (long long)L_ * L_,
        B_, nullptr, nullptr, DK_, 0.125f);

    // 5) softmax over last dim, in place in d_out attn region
    softmax_rows<<<H_ * B_ * L_, 256, 0, stream>>>(attn);

    // 6) attn @ V: per (h,b): attn fp32 [L,L] (cvt on load) @ vh_h [L,64] -> bf16
    gemm_wmma<128, 64, 128, false, true, false, false, false, true>
        <<<dim3(1, 8, H_ * B_), 128, 0, stream>>>(
        attn,  L_,       (long long)B_ * L_ * L_, (long long)L_ * L_,
        vh_bf, H_ * DV_, (long long)DV_, (long long)L_ * H_ * DV_,
        ao_bf, H_ * DV_, (long long)DV_, (long long)L_ * H_ * DV_,
        B_, nullptr, nullptr, L_, 1.0f);

    // 7) fc + bias + residual -> out1 fp32
    gemm_wmma<128, 128, 256, false, false, true, false, true, false>
        <<<dim3(8, 32, 1), 256, 0, stream>>>(
        ao_bf, H_ * DV_, 0, 0, fc_bf, D_, 0, 0, out1, D_, 0, 0,
        1, fc_b, q_res, H_ * DV_, 1.0f);

    // 8) SLN2: out1 -> h2_bf
    sln_kernel<false><<<B_ * L_, 256, 0, stream>>>(out1, z, ln2_w, ln2_b, g2, b2,
                                                   nullptr, h2_bf);

    // 9) MLP1 + bias + exact GELU -> bf16 [4096,4096]
    gemm_wmma<128, 128, 256, false, false, true, true, false, true>
        <<<dim3(32, 32, 1), 256, 0, stream>>>(
        h2_bf, D_, 0, 0, mlp1_bf, HID_, 0, 0, hid_bf, HID_, 0, 0,
        1, mlp1_b, nullptr, D_, 1.0f);

    // 10) MLP2 + bias + residual(out1) -> d_out fp32
    gemm_wmma<128, 128, 256, false, false, true, false, true, false>
        <<<dim3(8, 32, 1), 256, 0, stream>>>(
        hid_bf, HID_, 0, 0, mlp2_bf, D_, 0, 0, out_f, D_, 0, 0,
        1, mlp2_b, out1, HID_, 1.0f);
}
